// MultiHeadAttention_17995912970874
// MI455X (gfx1250) — matrix operand order repaired and checked by reading, not compiled, not run on hardware
//
#include <hip/hip_runtime.h>
#include <hip/hip_bf16.h>

// ---------------- CDNA5 WMMA types & helpers ----------------
typedef __attribute__((ext_vector_type(16))) __bf16 v16bf;
typedef __attribute__((ext_vector_type(8)))  __bf16 v8bf;
typedef __attribute__((ext_vector_type(8)))  float  v8f;
typedef int v4i __attribute__((vector_size(16)));
typedef __attribute__((address_space(1))) v4i* gptr_v4i;
typedef __attribute__((address_space(3))) v4i* lptr_v4i;

#define BATCH 2
#define SEQ   2048
#define DIM   768
#define HEADS 12
#define DK    64
#define MROWS (BATCH * SEQ)   // 4096

// ---- async global->LDS (ASYNCcnt) path, guarded ----
#if defined(__has_builtin)
#if __has_builtin(__builtin_amdgcn_global_load_async_to_lds_b128) && \
    __has_builtin(__builtin_amdgcn_s_wait_asynccnt)
#define HAVE_ASYNC_LDS 1
#endif
#endif
#ifndef HAVE_ASYNC_LDS
#define HAVE_ASYNC_LDS 0
#endif

__device__ __forceinline__ void copy_b128_to_lds(void* lds_dst, const void* gsrc) {
#if HAVE_ASYNC_LDS
    __builtin_amdgcn_global_load_async_to_lds_b128(
        (gptr_v4i)gsrc, (lptr_v4i)lds_dst, 0, 0);
#else
    *(uint4*)lds_dst = *(const uint4*)gsrc;
#endif
}
__device__ __forceinline__ void async_copy_wait() {
#if HAVE_ASYNC_LDS
    __builtin_amdgcn_s_wait_asynccnt(0);
#endif
}

__device__ __forceinline__ unsigned short f32_to_bfu(float f) {
    union { float f; unsigned u; } c; c.f = f;
    unsigned r = (c.u + 0x7FFFu + ((c.u >> 16) & 1u)) >> 16;
    return (unsigned short)r;
}
__device__ __forceinline__ unsigned int pack_bf2(float a, float b) {
    return (unsigned)f32_to_bfu(a) | ((unsigned)f32_to_bfu(b) << 16);
}
__device__ __forceinline__ v16bf cat8(v8bf lo, v8bf hi) {
    return __builtin_shufflevector(lo, hi, 0,1,2,3,4,5,6,7,8,9,10,11,12,13,14,15);
}
// B-fragment: 16 contiguous bf16 (32B) -> 2x ds_load_b128
__device__ __forceinline__ v16bf lds_load_v16(const unsigned short* p) {
    return cat8(*(const v8bf*)p, *(const v8bf*)(p + 8));
}
// A-fragment (16x32 16-bit): 16 contiguous k at 16*half (matches B)
__device__ __forceinline__ v16bf lds_load_afrag(const unsigned short* rowp, int half) {
    return cat8(*(const v8bf*)(rowp + 16 * half),
                *(const v8bf*)(rowp + 16 * half + 8));
}
__device__ __forceinline__ v8f wmma_bf16(v16bf a, v16bf b, v8f c) {
    return __builtin_amdgcn_wmma_f32_16x16x32_bf16(
        false, a, false, b, (short)0, c, false, false);
}

// ============================================================
// Kernel 1: QKV projection GEMM (64x64 tile, 4 waves)
//   Out[(b*H+h)*S+s][d] = scale * (X @ W)[m][h*64+d]   (bf16)
// ============================================================
__global__ __launch_bounds__(128)
void mha_proj_gemm_kernel(const float* __restrict__ X,
                          const float* __restrict__ W,
                          unsigned short* __restrict__ Out,
                          float scale)
{
    __shared__ unsigned short As[64][40];   // A tile 64x32 bf16, padded rows
    __shared__ unsigned short Wt[64][40];   // W tile transposed: Wt[n][k]

    const int tid  = threadIdx.x;
    const int wave = tid >> 5;
    const int lane = tid & 31;
    const int half = lane >> 4;
    const int l16  = lane & 15;
    const int m0 = blockIdx.x * 64;
    const int n0 = blockIdx.y * 64;

    v8f acc[4] = {};

    for (int k0 = 0; k0 < DIM; k0 += 32) {
        if (k0 + 32 < DIM)
            __builtin_prefetch(&W[(size_t)(k0 + 32) * DIM + n0], 0, 3);

        // stage A: 64x32, fp32->bf16, packed b128 stores
        for (int task = tid; task < 256; task += 128) {
            int r = task >> 2, cg = task & 3;
            const float* src = &X[(size_t)(m0 + r) * DIM + k0 + cg * 8];
            float4 f0 = *(const float4*)(src);
            float4 f1 = *(const float4*)(src + 4);
            uint4 pk;
            pk.x = pack_bf2(f0.x, f0.y); pk.y = pack_bf2(f0.z, f0.w);
            pk.z = pack_bf2(f1.x, f1.y); pk.w = pack_bf2(f1.z, f1.w);
            *(uint4*)&As[r][cg * 8] = pk;
        }
        // stage W transposed: Wt[n][k] (coalesced along n), packed b128
        for (int task = tid; task < 256; task += 128) {
            int n = task & 63, kg = task >> 6;
            float v[8];
            #pragma unroll
            for (int k = 0; k < 8; ++k)
                v[k] = W[(size_t)(k0 + kg * 8 + k) * DIM + n0 + n];
            uint4 pk;
            pk.x = pack_bf2(v[0], v[1]); pk.y = pack_bf2(v[2], v[3]);
            pk.z = pack_bf2(v[4], v[5]); pk.w = pack_bf2(v[6], v[7]);
            *(uint4*)&Wt[n][kg * 8] = pk;
        }
        __syncthreads();

        v16bf a = lds_load_afrag(&As[wave * 16 + l16][0], half);
        #pragma unroll
        for (int j = 0; j < 4; ++j) {
            v16bf b = lds_load_v16(&Wt[j * 16 + l16][16 * half]);
            acc[j] = wmma_bf16(a, b, acc[j]);
        }
        __syncthreads();
    }

    // scatter-store to [B*H, S, DK] bf16 (coalesced along d)
    #pragma unroll
    for (int j = 0; j < 4; ++j) {
        #pragma unroll
        for (int r = 0; r < 8; ++r) {
            int m = m0 + wave * 16 + 8 * half + r;
            int n = n0 + j * 16 + l16;
            int bb = m >> 11, s = m & 2047;
            int h  = n >> 6,  d = n & 63;
            Out[(((size_t)bb * HEADS + h) * SEQ + s) * DK + d] =
                f32_to_bfu(acc[j][r] * scale);
        }
    }
}

// ============================================================
// Kernel 2: flash attention, S^T formulation
//   S^T = K @ Q^T  (rows=key in regs, cols=q in lanes)
//   -> per-lane softmax stats, contiguous P chunks per lane
// ============================================================
__global__ __launch_bounds__(128)
void mha_attn_kernel(const unsigned short* __restrict__ Qb,
                     const unsigned short* __restrict__ Kb,
                     const unsigned short* __restrict__ Vb,
                     unsigned short* __restrict__ Cb)
{
    __shared__ unsigned short Qs[64][72];
    __shared__ unsigned short Ks[64][72];
    __shared__ unsigned short Vt[64][72];   // V transposed: Vt[d][key]
    __shared__ unsigned short Ps[64][72];   // P row-major [q][key]

    const int tid  = threadIdx.x;
    const int wave = tid >> 5;
    const int lane = tid & 31;
    const int half = lane >> 4;
    const int l16  = lane & 15;
    const int q0 = blockIdx.x * 64;
    const int bh = blockIdx.y;                 // 0..23
    const int bb = bh / HEADS, hh = bh % HEADS;

    const unsigned short* Qh = Qb + (size_t)bh * SEQ * DK;
    const unsigned short* Kh = Kb + (size_t)bh * SEQ * DK;
    const unsigned short* Vh = Vb + (size_t)bh * SEQ * DK;

    // stage Q tile once (64x64 bf16) into padded rows (async -> LDS)
    {
        const uint4* src = (const uint4*)(Qh + (size_t)q0 * DK);
        for (int i = tid; i < 512; i += 128) {
            int r = i >> 3, c = (i & 7) * 8;
            copy_b128_to_lds(&Qs[r][c], &src[i]);
        }
    }
    async_copy_wait();
    __syncthreads();

    // Q^T B-fragments: lane n = q, k = d contiguous
    v16bf qbf[2];
    #pragma unroll
    for (int s = 0; s < 2; ++s)
        qbf[s] = lds_load_v16(&Qs[wave * 16 + l16][32 * s + 16 * half]);

    v8f ctx[4] = {};
    float mcur = -1e30f, lcur = 0.0f;   // per lane: stats for q = wave*16+l16

    for (int t = 0; t < SEQ / 64; ++t) {
        const int kbase = t * 64;
        __syncthreads();    // previous iter done with Ks/Vt
        // stage K tile (natural [key][d] layout, async -> LDS)
        {
            const uint4* src = (const uint4*)(Kh + (size_t)kbase * DK);
            for (int i = tid; i < 512; i += 128) {
                int r = i >> 3, c = (i & 7) * 8;
                copy_b128_to_lds(&Ks[r][c], &src[i]);
            }
        }
        // stage V transposed: Vt[d][key] (coalesced along d), packed b128
        for (int task = tid; task < 256; task += 128) {
            int d = task & 63, kg = task >> 6;
            unsigned v[8];
            #pragma unroll
            for (int k2 = 0; k2 < 8; ++k2)
                v[k2] = Vh[(size_t)(kbase + kg * 8 + k2) * DK + d];
            uint4 pk;
            pk.x = v[0] | (v[1] << 16); pk.y = v[2] | (v[3] << 16);
            pk.z = v[4] | (v[5] << 16); pk.w = v[6] | (v[7] << 16);
            *(uint4*)&Vt[d][kg * 8] = pk;
        }
        async_copy_wait();
        __syncthreads();

        // S^T = K @ Q^T : frag j holds keys j*16.. x 16 q's
        v8f st[4];
        #pragma unroll
        for (int j = 0; j < 4; ++j) {
            v8f z = {};
            #pragma unroll
            for (int s = 0; s < 2; ++s) {
                v16bf ka = lds_load_afrag(&Ks[j * 16 + l16][32 * s], half);
                z = wmma_bf16(ka, qbf[s], z);
            }
            st[j] = z;
        }

        // online softmax: per-lane over 32 values + one shfl_xor(16)
        float vmax = -1e30f;
        #pragma unroll
        for (int j = 0; j < 4; ++j)
            #pragma unroll
            for (int r = 0; r < 8; ++r)
                vmax = fmaxf(vmax, st[j][r]);
        vmax = fmaxf(vmax, __shfl_xor(vmax, 16));
        float newm  = fmaxf(mcur, vmax);
        float alpha = __expf(mcur - newm);
        float rs = 0.0f;
        float pv[4][8];
        #pragma unroll
        for (int j = 0; j < 4; ++j)
            #pragma unroll
            for (int r = 0; r < 8; ++r) {
                float p = __expf(st[j][r] - newm);
                pv[j][r] = p;
                rs += p;
            }
        rs += __shfl_xor(rs, 16);
        lcur = lcur * alpha + rs;
        mcur = newm;

        // rescale ctx (C-layout rows q=r+8*half): broadcast alpha per row
        #pragma unroll
        for (int r = 0; r < 8; ++r) {
            float ar = __shfl(alpha, r + 8 * half);
            #pragma unroll
            for (int j = 0; j < 4; ++j) ctx[j][r] *= ar;
        }

        // store P: lane owns 8 contiguous keys per frag -> b128 stores
        #pragma unroll
        for (int j = 0; j < 4; ++j) {
            uint4 pk;
            pk.x = pack_bf2(pv[j][0], pv[j][1]);
            pk.y = pack_bf2(pv[j][2], pv[j][3]);
            pk.z = pack_bf2(pv[j][4], pv[j][5]);
            pk.w = pack_bf2(pv[j][6], pv[j][7]);
            *(uint4*)&Ps[wave * 16 + l16][j * 16 + 8 * half] = pk;
        }

        // ctx += P @ V : A = P (row-major), B = V^T rows (key-contiguous)
        #pragma unroll
        for (int s = 0; s < 2; ++s) {
            v16bf pa = lds_load_afrag(&Ps[wave * 16 + l16][32 * s], half);
            #pragma unroll
            for (int j = 0; j < 4; ++j) {
                v16bf vb = lds_load_v16(&Vt[j * 16 + l16][32 * s + 16 * half]);
                ctx[j] = wmma_bf16(pa, vb, ctx[j]);
            }
        }
    }

    // normalize (broadcast l per C-row) and store [B,S,H*dk] bf16
    float lr[8];
    #pragma unroll
    for (int r = 0; r < 8; ++r) lr[r] = __shfl(lcur, r + 8 * half);
    #pragma unroll
    for (int j = 0; j < 4; ++j)
        #pragma unroll
        for (int r = 0; r < 8; ++r) {
            int srow = q0 + wave * 16 + 8 * half + r;
            int col  = j * 16 + l16;
            Cb[((size_t)bb * SEQ + srow) * DIM + hh * DK + col] =
                f32_to_bfu(ctx[j][r] / lr[r]);
        }
}

// ============================================================
// Kernel 3: output projection + residual (f32 out)
// ============================================================
__global__ __launch_bounds__(128)
void mha_oproj_kernel(const unsigned short* __restrict__ Cbf,
                      const float* __restrict__ Wo,
                      const float* __restrict__ Res,
                      float* __restrict__ O)
{
    __shared__ unsigned short As[64][40];
    __shared__ unsigned short Wt[64][40];

    const int tid  = threadIdx.x;
    const int wave = tid >> 5;
    const int lane = tid & 31;
    const int half = lane >> 4;
    const int l16  = lane & 15;
    const int m0 = blockIdx.x * 64;
    const int n0 = blockIdx.y * 64;

    v8f acc[4] = {};

    for (int k0 = 0; k0 < DIM; k0 += 32) {
        if (k0 + 32 < DIM)
            __builtin_prefetch(&Wo[(size_t)(k0 + 32) * DIM + n0], 0, 3);

        // A tile already bf16: async b128 copies straight into LDS
        for (int task = tid; task < 256; task += 128) {
            int r = task >> 2, cg = task & 3;
            copy_b128_to_lds(&As[r][cg * 8],
                             &Cbf[(size_t)(m0 + r) * DIM + k0 + cg * 8]);
        }
        for (int task = tid; task < 256; task += 128) {
            int n = task & 63, kg = task >> 6;
            float v[8];
            #pragma unroll
            for (int k = 0; k < 8; ++k)
                v[k] = Wo[(size_t)(k0 + kg * 8 + k) * DIM + n0 + n];
            uint4 pk;
            pk.x = pack_bf2(v[0], v[1]); pk.y = pack_bf2(v[2], v[3]);
            pk.z = pack_bf2(v[4], v[5]); pk.w = pack_bf2(v[6], v[7]);
            *(uint4*)&Wt[n][kg * 8] = pk;
        }
        async_copy_wait();
        __syncthreads();

        v16bf a = lds_load_afrag(&As[wave * 16 + l16][0], half);
        #pragma unroll
        for (int j = 0; j < 4; ++j) {
            v16bf b = lds_load_v16(&Wt[j * 16 + l16][16 * half]);
            acc[j] = wmma_bf16(a, b, acc[j]);
        }
        __syncthreads();
    }

    #pragma unroll
    for (int j = 0; j < 4; ++j)
        #pragma unroll
        for (int r = 0; r < 8; ++r) {
            int m = m0 + wave * 16 + 8 * half + r;
            int n = n0 + j * 16 + l16;
            size_t idx = (size_t)m * DIM + n;
            O[idx] = acc[j][r] + Res[idx];
        }
}

// ============================================================
// Kernel 4: LayerNorm(768) per row (weight=1, bias=0)
// ============================================================
__global__ __launch_bounds__(256)
void mha_ln_kernel(const float* __restrict__ X, float* __restrict__ Out)
{
    const int row  = blockIdx.x;
    const int tid  = threadIdx.x;
    const int lane = tid & 31;
    const int wv   = tid >> 5;

    __shared__ float sred[8];
    __shared__ float smu;
    __shared__ float srstd;

    const size_t base = (size_t)row * DIM;
    float x0 = X[base + tid];
    float x1 = X[base + 256 + tid];
    float x2 = X[base + 512 + tid];

    float s = x0 + x1 + x2;
    #pragma unroll
    for (int m = 1; m < 32; m <<= 1) s += __shfl_xor(s, m);
    if (lane == 0) sred[wv] = s;
    __syncthreads();
    if (tid == 0) {
        float t = 0.0f;
        #pragma unroll
        for (int i = 0; i < 8; ++i) t += sred[i];
        smu = t * (1.0f / (float)DIM);
    }
    __syncthreads();

    float mu = smu;
    float d0 = x0 - mu, d1 = x1 - mu, d2 = x2 - mu;
    float q = d0 * d0 + d1 * d1 + d2 * d2;
    #pragma unroll
    for (int m = 1; m < 32; m <<= 1) q += __shfl_xor(q, m);
    if (lane == 0) sred[wv] = q;
    __syncthreads();
    if (tid == 0) {
        float t = 0.0f;
        #pragma unroll
        for (int i = 0; i < 8; ++i) t += sred[i];
        srstd = rsqrtf(t * (1.0f / (float)DIM) + 1e-5f);
    }
    __syncthreads();

    float rs = srstd;
    Out[base + tid]       = d0 * rs;
    Out[base + 256 + tid] = d1 * rs;
    Out[base + 512 + tid] = d2 * rs;
}

// ============================================================
// Host launch
// ============================================================
extern "C" void kernel_launch(void* const* d_in, const int* in_sizes, int n_in,
                              void* d_out, int out_size, void* d_ws, size_t ws_size,
                              hipStream_t stream) {
    const float* inQ = (const float*)d_in[0];
    const float* inK = (const float*)d_in[1];
    const float* inV = (const float*)d_in[2];
    const float* Wq  = (const float*)d_in[3];
    const float* Wk  = (const float*)d_in[4];
    const float* Wv  = (const float*)d_in[5];
    const float* Wo  = (const float*)d_in[6];

    const size_t nElems = (size_t)MROWS * DIM;   // 4096*768
    unsigned short* Qb = (unsigned short*)d_ws;
    unsigned short* Kb = Qb + nElems;
    unsigned short* Vb = Kb + nElems;
    unsigned short* Cb = Vb + nElems;
    float*          Ob = (float*)(Cb + nElems);

    dim3 gemmGrid(MROWS / 64, DIM / 64);   // 64 x 12
    dim3 gemmBlk(128);

    const float qscale = 0.125f;   // 1/sqrt(64)
    mha_proj_gemm_kernel<<<gemmGrid, gemmBlk, 0, stream>>>(inQ, Wq, Qb, qscale);
    mha_proj_gemm_kernel<<<gemmGrid, gemmBlk, 0, stream>>>(inK, Wk, Kb, 1.0f);
    mha_proj_gemm_kernel<<<gemmGrid, gemmBlk, 0, stream>>>(inV, Wv, Vb, 1.0f);

    mha_attn_kernel<<<dim3(SEQ / 64, BATCH * HEADS), gemmBlk, 0, stream>>>(Qb, Kb, Vb, Cb);

    mha_oproj_kernel<<<gemmGrid, gemmBlk, 0, stream>>>(Cb, Wo, inQ, Ob);

    mha_ln_kernel<<<dim3(MROWS), dim3(256), 0, stream>>>(Ob, (float*)d_out);
}
